// Net_5128190952056
// MI455X (gfx1250) — matrix operand order repaired and checked by reading, not compiled, not run on hardware
//
#include <hip/hip_runtime.h>
#include <hip/hip_bf16.h>
#include <math.h>

// ---- problem constants (from reference) ----
#define BS    32
#define GRID_N 2704
#define NA    3
#define NCH   85
#define VDIM  1024
#define HD    512
#define SEL   32
#define EPSV  1e-8f
#define MROWS (BS*SEL)          // 1024 fused rows

typedef __attribute__((ext_vector_type(16))) _Float16 v16h;
typedef __attribute__((ext_vector_type(8)))  float    v8f;

// ---------- WMMA fragment loaders (CDNA5 wave32 layouts, ISA §7.12.2) ----------
// A 16x32 f16, row-major [M][K]: lane<16 -> M=lane, K in {k0..k0+7, k0+16..k0+23}
//                                lane>=16 -> M=lane-16, K in {k0+8..15, k0+24..31}
__device__ __forceinline__ v16h load_a_frag(const _Float16* base, int ld,
                                            int m0, int k0, int lane) {
    int half = lane >> 4, r = lane & 15;
    const _Float16* p = base + (size_t)(m0 + r) * ld + k0 + half * 8;
    union { v16h v; uint4 u[2]; } U;
    U.u[0] = *(const uint4*)(p);
    U.u[1] = *(const uint4*)(p + 16);
    return U.v;
}
// B 32x16 f16, stored transposed [N][K]: lane<16 -> N=lane, K in {k0..k0+7, k0+16..k0+23}
//                                        lane>=16 -> N=lane-16, K in {k0+8..15, k0+24..31}
__device__ __forceinline__ v16h load_b_frag(const _Float16* baseT, int ld,
                                            int n0, int k0, int lane) {
    int half = lane >> 4, c = lane & 15;
    const _Float16* p = baseT + (size_t)(n0 + c) * ld + k0 + half * 8;
    union { v16h v; uint4 u[2]; } U;
    U.u[0] = *(const uint4*)(p);
    U.u[1] = *(const uint4*)(p + 16);
    return U.v;
}

// ---------- K1: per-batch top-32 on mean objectness, sorted ascending ----------
__global__ void topk_kernel(const float* __restrict__ boxes, int* __restrict__ idx_out) {
    __shared__ float sc[GRID_N];
    __shared__ float rv[256];
    __shared__ int   ri[256];
    __shared__ int   sel[SEL];
    const int b = blockIdx.x, t = threadIdx.x;
    const float inv3 = 1.0f / 3.0f;
    for (int g = t; g < GRID_N; g += 256) {
        const float* p = boxes + ((size_t)b * GRID_N + g) * (NA * NCH);
        sc[g] = (p[4] + p[NCH + 4] + p[2 * NCH + 4]) * inv3;   // uniform[0,1] means
    }
    __syncthreads();
    for (int it = 0; it < SEL; ++it) {
        float bv = -1.0f; int bi = GRID_N;
        for (int g = t; g < GRID_N; g += 256) {
            float v = sc[g];
            if (v > bv || (v == bv && g < bi)) { bv = v; bi = g; }
        }
        rv[t] = bv; ri[t] = bi;
        __syncthreads();
        for (int off = 128; off > 0; off >>= 1) {
            if (t < off) {
                float v2 = rv[t + off]; int i2 = ri[t + off];
                if (v2 > rv[t] || (v2 == rv[t] && i2 < ri[t])) { rv[t] = v2; ri[t] = i2; }
            }
            __syncthreads();
        }
        if (t == 0) { sel[it] = ri[0]; sc[ri[0]] = -1.0f; }
        __syncthreads();
    }
    if (t < SEL) {                      // rank-sort 32 distinct indices ascending
        int my = sel[t], rank = 0;
        for (int j = 0; j < SEL; ++j) rank += (sel[j] < my);
        idx_out[b * SEL + rank] = my;
    }
}

// ---------- K2: gather x_feat[b, :, idx] -> A1 f16 [MROWS][VDIM] ----------
__global__ void gather_kernel(const float* __restrict__ x_feat,
                              const int* __restrict__ idx,
                              _Float16* __restrict__ A1) {
    const int row = blockIdx.x;                // b*SEL + s
    const int b = row >> 5;
    const int g = idx[row];
    const float* src = x_feat + (size_t)b * VDIM * GRID_N + g;
    _Float16* dst = A1 + (size_t)row * VDIM;
    for (int v = threadIdx.x; v < VDIM; v += blockDim.x)
        dst[v] = (_Float16)src[(size_t)v * GRID_N];
}

// ---------- K3: W[K][N] f32 -> WT[N][K] f16 ----------
__global__ void transpose_f16_kernel(const float* __restrict__ W,
                                     _Float16* __restrict__ WT, int K, int N) {
    int i = blockIdx.x * blockDim.x + threadIdx.x;
    if (i < K * N) {
        int k = i / N, n = i % N;
        WT[(size_t)n * K + k] = (_Float16)W[i];
    }
}

// ---------- K4: lang_e = lang @ W_ts + b_ts  (32x512x512, scalar) ----------
__global__ void lang_kernel(const float* __restrict__ lang,
                            const float* __restrict__ W_ts,
                            const float* __restrict__ b_ts,
                            float* __restrict__ langE) {
    int i = blockIdx.x * blockDim.x + threadIdx.x;   // b*HD + h
    int b = i >> 9, h = i & (HD - 1);
    const float* l = lang + b * HD;
    float acc = b_ts[h];
    for (int k = 0; k < HD; ++k) acc = fmaf(l[k], W_ts[k * HD + h], acc);
    langE[i] = acc;
}

// ---------- K5: vis = A1 @ W_vs + b_vs  (WMMA f16 -> f32) ----------
__global__ void gemm1_kernel(const _Float16* __restrict__ A1,
                             const _Float16* __restrict__ WvsT,
                             const float* __restrict__ b_vs,
                             float* __restrict__ vis) {
    const int lane = threadIdx.x & 31;
    const int tile = blockIdx.x * 8 + (threadIdx.x >> 5);   // 2048 tiles
    const int m0 = (tile >> 5) << 4;                        // 64 row tiles
    const int n0 = (tile & 31) << 4;                        // 32 col tiles
    v8f acc = {};
    for (int k0 = 0; k0 < VDIM; k0 += 32) {
        v16h a = load_a_frag(A1,   VDIM, m0, k0, lane);
        v16h bm = load_b_frag(WvsT, VDIM, n0, k0, lane);
        acc = __builtin_amdgcn_wmma_f32_16x16x32_f16(false, a, false, bm,
                                                     (short)0, acc, false, false);
    }
    const int col  = n0 + (lane & 15);
    const int mrow = m0 + (lane >> 4) * 8;
    const float bv = b_vs[col];
#pragma unroll
    for (int v = 0; v < 8; ++v)
        vis[(size_t)(mrow + v) * HD + col] = acc[v] + bv;
}

// ---------- K6: gating log-softmax; emit A2 = vis*w0, A3 = tag*w1 (f16) ----------
__global__ void gate_kernel(const float* __restrict__ vis,
                            const float* __restrict__ tag,
                            const float* __restrict__ pos,
                            const float* __restrict__ W_soft,
                            const float* __restrict__ b_soft,
                            _Float16* __restrict__ A2,
                            _Float16* __restrict__ A3) {
    __shared__ float s0[256], s1[256];
    __shared__ float w0s, w1s;
    const int m = blockIdx.x, t = threadIdx.x;
    const float* vr = vis + (size_t)m * HD;
    const float* tr = tag + (size_t)m * HD;
    const float* pr = pos + (size_t)m * HD;
    float p0 = 0.f, p1 = 0.f;
    for (int h = t; h < HD; h += 256) {
        float x = vr[h] + tr[h] + pr[h];
        p0 = fmaf(x, W_soft[2 * h],     p0);
        p1 = fmaf(x, W_soft[2 * h + 1], p1);
    }
    s0[t] = p0; s1[t] = p1;
    __syncthreads();
    for (int off = 128; off > 0; off >>= 1) {
        if (t < off) { s0[t] += s0[t + off]; s1[t] += s1[t + off]; }
        __syncthreads();
    }
    if (t == 0) {
        float l0 = (s0[0] + b_soft[0]) * (1.0f / 0.03f);
        float l1 = (s1[0] + b_soft[1]) * (1.0f / 0.03f);
        float mx = fmaxf(l0, l1);
        float lse = mx + logf(expf(l0 - mx) + expf(l1 - mx));
        w0s = l0 - lse; w1s = l1 - lse;
    }
    __syncthreads();
    const float w0 = w0s, w1 = w1s;
    for (int h = t; h < HD; h += 256) {
        A2[(size_t)m * HD + h] = (_Float16)(vr[h] * w0);
        A3[(size_t)m * HD + h] = (_Float16)(tr[h] * w1);
    }
}

// ---------- K7: vis2 = A2@W_vs_pos + A3@W_tag + biases + pos (WMMA) ----------
__global__ void gemm2_kernel(const _Float16* __restrict__ A2,
                             const _Float16* __restrict__ A3,
                             const _Float16* __restrict__ WpT,
                             const _Float16* __restrict__ WtT,
                             const float* __restrict__ b_vs_pos,
                             const float* __restrict__ b_tag,
                             const float* __restrict__ pos,
                             float* __restrict__ vis2) {
    const int lane = threadIdx.x & 31;
    const int tile = blockIdx.x * 8 + (threadIdx.x >> 5);
    const int m0 = (tile >> 5) << 4;
    const int n0 = (tile & 31) << 4;
    v8f acc = {};
    for (int k0 = 0; k0 < HD; k0 += 32) {
        v16h a2 = load_a_frag(A2, HD, m0, k0, lane);
        v16h bp = load_b_frag(WpT, HD, n0, k0, lane);
        acc = __builtin_amdgcn_wmma_f32_16x16x32_f16(false, a2, false, bp,
                                                     (short)0, acc, false, false);
        v16h a3 = load_a_frag(A3, HD, m0, k0, lane);
        v16h bt = load_b_frag(WtT, HD, n0, k0, lane);
        acc = __builtin_amdgcn_wmma_f32_16x16x32_f16(false, a3, false, bt,
                                                     (short)0, acc, false, false);
    }
    const int col  = n0 + (lane & 15);
    const int mrow = m0 + (lane >> 4) * 8;
    const float bc = b_vs_pos[col] + b_tag[col];
#pragma unroll
    for (int v = 0; v < 8; ++v) {
        size_t o = (size_t)(mrow + v) * HD + col;
        vis2[o] = acc[v] + bc + pos[o];
    }
}

// ---------- K8: cosine sim, argmax, box decode; write outputs ----------
__global__ void head_kernel(const float* __restrict__ vis2,
                            const float* __restrict__ langE,
                            const int* __restrict__ idx,
                            const float* __restrict__ boxes,
                            float* __restrict__ out) {
    __shared__ float red[256];
    __shared__ float simbuf[SEL];
    __shared__ float lden_s;
    const int b = blockIdx.x, t = threadIdx.x;
    const int lane = t & 31, wave = t >> 5;
    const float* le = langE + b * HD;
    float p = 0.f;
    for (int h = t; h < HD; h += 256) { float x = le[h]; p = fmaf(x, x, p); }
    red[t] = p;
    __syncthreads();
    for (int off = 128; off > 0; off >>= 1) {
        if (t < off) red[t] += red[t + off];
        __syncthreads();
    }
    if (t == 0) lden_s = sqrtf(red[0]) + EPSV;
    __syncthreads();
    const float lden = lden_s;
    for (int s = wave; s < SEL; s += 8) {
        const float* vr = vis2 + (size_t)(b * SEL + s) * HD;
        float dot = 0.f, nn = 0.f;
        for (int h = lane; h < HD; h += 32) {
            float x = vr[h];
            dot = fmaf(x, le[h], dot);
            nn  = fmaf(x, x, nn);
        }
        for (int m = 16; m > 0; m >>= 1) {
            dot += __shfl_xor(dot, m, 32);
            nn  += __shfl_xor(nn,  m, 32);
        }
        if (lane == 0) {
            float sim = dot / ((sqrtf(nn) + EPSV) * lden);
            simbuf[s] = sim;
            out[BS * 5 + b * SEL + s] = sim;     // sim section after box_pred (160)
        }
    }
    __syncthreads();
    if (t == 0) {
        int best = 0; float bv = simbuf[0];
        for (int s = 1; s < SEL; ++s)
            if (simbuf[s] > bv) { bv = simbuf[s]; best = s; }   // first-max = argmax
        const int g = idx[b * SEL + best];
        const float* bp = boxes + ((size_t)b * GRID_N + g) * (NA * NCH);
        float box[5] = {0, 0, 0, 0, 0};
        float bestObj = -1e30f;
        for (int a = 0; a < NA; ++a) {
            const float* q = bp + a * NCH;
            float x1 = q[0] - q[2] * 0.5f;
            float y1 = q[1] - q[3] * 0.5f;
            float obj = q[4];
            if (obj > bestObj) {
                bestObj = obj;
                box[0] = x1; box[1] = y1;
                box[2] = x1 + q[2]; box[3] = y1 + q[3]; box[4] = obj;
            }
        }
        for (int c = 0; c < 5; ++c) out[b * 5 + c] = box[c];
    }
}

extern "C" void kernel_launch(void* const* d_in, const int* in_sizes, int n_in,
                              void* d_out, int out_size, void* d_ws, size_t ws_size,
                              hipStream_t stream) {
    (void)in_sizes; (void)n_in; (void)out_size; (void)ws_size;
    const float* boxes    = (const float*)d_in[0];
    const float* x_feat   = (const float*)d_in[1];
    const float* tag_emb  = (const float*)d_in[2];
    const float* pos_emb  = (const float*)d_in[3];
    const float* lang     = (const float*)d_in[4];
    const float* W_vs     = (const float*)d_in[5];
    const float* b_vs     = (const float*)d_in[6];
    const float* W_ts     = (const float*)d_in[7];
    const float* b_ts     = (const float*)d_in[8];
    const float* W_vs_pos = (const float*)d_in[9];
    const float* b_vs_pos = (const float*)d_in[10];
    const float* W_tag    = (const float*)d_in[11];
    const float* b_tag    = (const float*)d_in[12];
    const float* W_soft   = (const float*)d_in[13];
    const float* b_soft   = (const float*)d_in[14];
    float* out = (float*)d_out;

    // workspace layout (all 256B-aligned sections)
    char* ws = (char*)d_ws;
    int*      idx   = (int*)ws;                              // 4 KB
    float*    langE = (float*)(ws + 4096);                   // 64 KB
    _Float16* WvsT  = (_Float16*)(ws + 77824);               // 1 MB   [512][1024]
    _Float16* WpT   = (_Float16*)(ws + 1126400);             // 512 KB [512][512]
    _Float16* WtT   = (_Float16*)(ws + 1650688);             // 512 KB [512][512]
    _Float16* A1    = (_Float16*)(ws + 2174976);             // 2 MB   [1024][1024]
    _Float16* A2    = (_Float16*)(ws + 4272128);             // 1 MB   [1024][512]
    _Float16* A3    = (_Float16*)(ws + 5320704);             // 1 MB   [1024][512]
    float*    vis   = (float*)(ws + 6369280);                // 2 MB   [1024][512]
    float*    vis2  = (float*)(ws + 8466432);                // 2 MB   [1024][512]

    topk_kernel<<<BS, 256, 0, stream>>>(boxes, idx);
    gather_kernel<<<MROWS, 256, 0, stream>>>(x_feat, idx, A1);
    transpose_f16_kernel<<<(VDIM * HD) / 256, 256, 0, stream>>>(W_vs, WvsT, VDIM, HD);
    transpose_f16_kernel<<<(HD * HD) / 256, 256, 0, stream>>>(W_vs_pos, WpT, HD, HD);
    transpose_f16_kernel<<<(HD * HD) / 256, 256, 0, stream>>>(W_tag, WtT, HD, HD);
    lang_kernel<<<(BS * HD) / 256, 256, 0, stream>>>(lang, W_ts, b_ts, langE);
    gemm1_kernel<<<(MROWS / 16) * (HD / 16) / 8, 256, 0, stream>>>(A1, WvsT, b_vs, vis);
    gate_kernel<<<MROWS, 256, 0, stream>>>(vis, tag_emb, pos_emb, W_soft, b_soft, A2, A3);
    gemm2_kernel<<<(MROWS / 16) * (HD / 16) / 8, 256, 0, stream>>>(
        A2, A3, WpT, WtT, b_vs_pos, b_tag, pos_emb, vis2);
    head_kernel<<<BS, 256, 0, stream>>>(vis2, langE, idx, boxes, out);
}
